// RNNDecoder_62998580298317
// MI455X (gfx1250) — matrix operand order repaired and checked by reading, not compiled, not run on hardware
//
#include <hip/hip_runtime.h>
#include <math.h>

// Problem dims (from reference): T=64 -> 63 decode steps, B=32, S=256, H=512, E=512, V=32000, L=2
#define NT 63
#define BB 32
#define SS 256
#define HH 512
#define EE 512
#define VV 32000
#define NG 2048   // 4*H gate width

typedef __attribute__((ext_vector_type(16))) __bf16        v16bf;
typedef __attribute__((ext_vector_type(8)))  float         v8f;
typedef __attribute__((ext_vector_type(2)))  float         v2f;
typedef __attribute__((ext_vector_type(8)))  unsigned int  v8u;

__device__ __forceinline__ float sigm(float x) { return 1.0f / (1.0f + expf(-x)); }

__device__ __forceinline__ unsigned short f2bf(float f) {
    unsigned int u = __builtin_bit_cast(unsigned int, f);
    u += 0x7fffu + ((u >> 16) & 1u);          // round-to-nearest-even
    return (unsigned short)(u >> 16);
}

// ---------------- xcat = [emb_table[tgt_t], input_feed]  (B x (E+H)) ----------------
__global__ __launch_bounds__(256) void xcat_kernel(
    const int* __restrict__ tgt_t, const float* __restrict__ emb,
    const float* __restrict__ feed, float* __restrict__ xc)
{
    int idx = blockIdx.x * 256 + threadIdx.x;       // B*1024 = 32768
    int b = idx >> 10, k = idx & 1023;
    float v = (k < EE) ? emb[(size_t)tgt_t[b] * EE + k] : feed[b * HH + (k - EE)];
    xc[idx] = v;
}

// ---------------- generic fused gate GEMM via V_WMMA_F32_16X16X4_F32 (full fp32) ----------
// out(32 x 2048) = A1(32xK1) @ W1^T(K1x2048) + A2(32xK2) @ W2^T(K2x2048) + bias
// Block = 128 threads (4 waves); wave covers 16 output cols, two 16-row M tiles.
// fp32 A 16x4 layout: lane m (=lane%16), VGPR j -> K = 2*(lane/16) + j. B symmetric.
template<int K1, int K2>
__device__ __forceinline__ void gates_seg(
    const float* __restrict__ A, const float* __restrict__ W, int K,
    int mrow, int half, int ncol, v8f& acc0, v8f& acc1)
{
    const float* ar0 = A + (size_t)mrow * K;
    const float* ar1 = A + (size_t)(mrow + 16) * K;
    const float* wr  = W + (size_t)ncol * K;
    for (int kk = 0; kk < K; kk += 4) {
        int ko = kk + half * 2;
        v2f a0, a1, bb;
        a0[0] = ar0[ko]; a0[1] = ar0[ko + 1];
        a1[0] = ar1[ko]; a1[1] = ar1[ko + 1];
        bb[0] = wr[ko];  bb[1] = wr[ko + 1];
        acc0 = __builtin_amdgcn_wmma_f32_16x16x4_f32(false, a0, false, bb, (short)0, acc0, false, false);
        acc1 = __builtin_amdgcn_wmma_f32_16x16x4_f32(false, a1, false, bb, (short)0, acc1, false, false);
    }
}

template<int K1, int K2>
__global__ __launch_bounds__(128) void gates_wmma_kernel(
    const float* __restrict__ A1, const float* __restrict__ W1,
    const float* __restrict__ A2, const float* __restrict__ W2,
    const float* __restrict__ bias, float* __restrict__ out)
{
    int lane = threadIdx.x & 31;
    int wave = threadIdx.x >> 5;
    int half = lane >> 4;
    int mrow = lane & 15;
    int ncol = blockIdx.x * 64 + wave * 16 + mrow;   // output gate column (0..2047)
    v8f acc0 = {}; v8f acc1 = {};
    gates_seg<K1, K2>(A1, W1, K1, mrow, half, ncol, acc0, acc1);
    gates_seg<K1, K2>(A2, W2, K2, mrow, half, ncol, acc0, acc1);
    float bg = bias[ncol];
    #pragma unroll
    for (int r = 0; r < 8; ++r) {
        int brow = half * 8 + r;                     // C layout: VGPR r -> M = half*8 + r, N = mrow
        out[(size_t)brow * NG + ncol] = acc0[r] + bg;
        out[(size_t)(brow + 16) * NG + ncol] = acc1[r] + bg;
    }
}

// ---------------- LSTM cell 0 elementwise, in-place h0/c0 ----------------
__global__ __launch_bounds__(256) void cell0_kernel(
    const float* __restrict__ g, float* __restrict__ h, float* __restrict__ c)
{
    int idx = blockIdx.x * 256 + threadIdx.x;       // B*H
    int b = idx >> 9, hh = idx & 511;
    const float* gr = g + (size_t)b * NG;
    float gi = gr[hh], gf = gr[HH + hh], gg = gr[2 * HH + hh], go = gr[3 * HH + hh];
    float cn = sigm(gf) * c[idx] + sigm(gi) * tanhf(gg);
    c[idx] = cn;
    h[idx] = sigm(go) * tanhf(cn);
}

// ---------------- LSTM cell 1 + q = h1' @ Wa (fused per-batch block) ----------------
__global__ __launch_bounds__(512) void cell1q_kernel(
    const float* __restrict__ g, float* __restrict__ h1, float* __restrict__ c1,
    const float* __restrict__ Wa, float* __restrict__ q)
{
    int b = blockIdx.x, tid = threadIdx.x;          // 32 blocks x 512
    __shared__ float hs[HH];
    const float* gr = g + (size_t)b * NG;
    float gi = gr[tid], gf = gr[HH + tid], gg = gr[2 * HH + tid], go = gr[3 * HH + tid];
    int idx = b * HH + tid;
    float cn = sigm(gf) * c1[idx] + sigm(gi) * tanhf(gg);
    float hn = sigm(go) * tanhf(cn);
    c1[idx] = cn; h1[idx] = hn; hs[tid] = hn;
    __syncthreads();
    float acc = 0.f;                                 // q[b,k] = sum_h h1[b,h]*Wa[h,k]
    #pragma unroll 4
    for (int h = 0; h < HH; ++h) acc += hs[h] * Wa[(size_t)h * HH + tid];
    q[idx] = acc;
}

// ---------------- align + masked softmax + context ----------------
__global__ __launch_bounds__(256) void attn_kernel(
    const float* __restrict__ q, const float* __restrict__ mb,
    const int* __restrict__ lens, float* __restrict__ attn_out, float* __restrict__ ctx)
{
    int b = blockIdx.x, tid = threadIdx.x;          // 32 blocks x 256 (tid = s)
    __shared__ float qs[HH];
    __shared__ float red[SS];
    __shared__ float pa[SS];
    qs[tid] = q[b * HH + tid]; qs[tid + 256] = q[b * HH + 256 + tid];
    __syncthreads();
    const float* mrow = mb + ((size_t)tid * BB + b) * HH;
    float al = 0.f;
    #pragma unroll 4
    for (int k = 0; k < HH; ++k) al += qs[k] * mrow[k];
    if (tid >= lens[b]) al = -1.0e9f;
    red[tid] = al; __syncthreads();
    for (int o = 128; o > 0; o >>= 1) { if (tid < o) red[tid] = fmaxf(red[tid], red[tid + o]); __syncthreads(); }
    float mx = red[0]; __syncthreads();
    float e = expf(al - mx);
    red[tid] = e; __syncthreads();
    for (int o = 128; o > 0; o >>= 1) { if (tid < o) red[tid] += red[tid + o]; __syncthreads(); }
    float p = e / red[0];
    pa[tid] = p;
    attn_out[b * SS + tid] = p;
    __syncthreads();
    for (int hh = tid; hh < HH; hh += 256) {
        float acc = 0.f;
        for (int s = 0; s < SS; ++s) acc += pa[s] * mb[((size_t)s * BB + b) * HH + hh];
        ctx[b * HH + hh] = acc;
    }
}

// ---------------- dec_out = tanh([ctx, h1] @ Wc^T); write feed(fp32) + bf16 copy ----------------
__global__ __launch_bounds__(256) void dec_kernel(
    const float* __restrict__ ctx, const float* __restrict__ h1,
    const float* __restrict__ Wc, float* __restrict__ feed, unsigned short* __restrict__ decbf)
{
    int idx = blockIdx.x * 256 + threadIdx.x;       // B*H
    int b = idx >> 9, j = idx & 511;
    const float* wc = Wc + (size_t)j * (2 * HH);
    const float* cr = ctx + b * HH;
    const float* hr = h1 + b * HH;
    float acc = 0.f;
    #pragma unroll 4
    for (int k = 0; k < HH; ++k) acc += cr[k] * wc[k];
    #pragma unroll 4
    for (int k = 0; k < HH; ++k) acc += hr[k] * wc[HH + k];
    float d = tanhf(acc);
    feed[idx] = d;
    decbf[idx] = f2bf(d);
}

// ---------------- one-time Wgen fp32 -> bf16 (row-major preserved) ----------------
__global__ __launch_bounds__(256) void cvt_wgen_kernel(
    const float* __restrict__ src, unsigned short* __restrict__ dst)
{
    int idx = blockIdx.x * 256 + threadIdx.x;       // V*H/4 = 4,096,000 threads
    const float4 v = ((const float4*)src)[idx];
    uint2 u;
    u.x = (unsigned int)f2bf(v.x) | ((unsigned int)f2bf(v.y) << 16);
    u.y = (unsigned int)f2bf(v.z) | ((unsigned int)f2bf(v.w) << 16);
    ((uint2*)dst)[idx] = u;
}

// ---------------- logits GEMM via v_wmma_f32_16x16x32_bf16, pre-converted Wgen ------------
__global__ __launch_bounds__(128) void wgen_wmma_fast(
    const unsigned short* __restrict__ adec, const unsigned short* __restrict__ wgenbf,
    const float* __restrict__ bgen, float* __restrict__ out)
{
    int lane = threadIdx.x & 31;
    int wave = threadIdx.x >> 5;
    int half = lane >> 4;
    int mrow = lane & 15;
    int vcol = blockIdx.x * 64 + wave * 16 + mrow;
    const unsigned int* a32 = (const unsigned int*)adec;
    const unsigned int* b32 = (const unsigned int*)wgenbf;
    size_t brow = ((size_t)vcol * HH) >> 1;

    v8f acc0 = {}; v8f acc1 = {};
    for (int kk = 0; kk < HH; kk += 32) {
        int abase0 = ((mrow * HH + kk) >> 1) + half * 4;          // A pairs: 2 x b128
        int abase1 = (((mrow + 16) * HH + kk) >> 1) + half * 4;
        size_t bbase = brow + (kk >> 1) + half * 4;               // B pairs: 2 x b128
        v8u apk0, apk1, bpk;
        #pragma unroll
        for (int j = 0; j < 4; ++j) {
            apk0[j]     = a32[abase0 + j];
            apk0[j + 4] = a32[abase0 + 8 + j];
            apk1[j]     = a32[abase1 + j];
            apk1[j + 4] = a32[abase1 + 8 + j];
        }
        #pragma unroll
        for (int j = 0; j < 8; ++j) bpk[j] = b32[bbase + j + (j >> 2) * 4];
        v16bf a0 = __builtin_bit_cast(v16bf, apk0);
        v16bf a1 = __builtin_bit_cast(v16bf, apk1);
        v16bf bf = __builtin_bit_cast(v16bf, bpk);
        acc0 = __builtin_amdgcn_wmma_f32_16x16x32_bf16(false, a0, false, bf, (short)0, acc0, false, false);
        acc1 = __builtin_amdgcn_wmma_f32_16x16x32_bf16(false, a1, false, bf, (short)0, acc1, false, false);
    }
    float bg = bgen[vcol];
    #pragma unroll
    for (int r = 0; r < 8; ++r) {
        int brw = half * 8 + r;
        out[(size_t)brw * VV + vcol] = acc0[r] + bg;
        out[(size_t)(brw + 16) * VV + vcol] = acc1[r] + bg;
    }
}

// ---------------- fallback: same GEMM with on-the-fly fp32->bf16 of Wgen ----------------
__global__ __launch_bounds__(128) void wgen_wmma_fly(
    const unsigned short* __restrict__ adec, const float* __restrict__ Wgen,
    const float* __restrict__ bgen, float* __restrict__ out)
{
    int lane = threadIdx.x & 31;
    int wave = threadIdx.x >> 5;
    int half = lane >> 4;
    int mrow = lane & 15;
    int vcol = blockIdx.x * 64 + wave * 16 + mrow;
    const unsigned int* a32 = (const unsigned int*)adec;
    const float* wrow = Wgen + (size_t)vcol * HH;

    v8f acc0 = {}; v8f acc1 = {};
    for (int kk = 0; kk < HH; kk += 32) {
        int abase0 = ((mrow * HH + kk) >> 1) + half * 4;
        int abase1 = (((mrow + 16) * HH + kk) >> 1) + half * 4;
        v8u apk0, apk1, bpk;
        #pragma unroll
        for (int j = 0; j < 4; ++j) {
            apk0[j]     = a32[abase0 + j];
            apk0[j + 4] = a32[abase0 + 8 + j];
            apk1[j]     = a32[abase1 + j];
            apk1[j + 4] = a32[abase1 + 8 + j];
        }
        #pragma unroll
        for (int j = 0; j < 8; ++j) {
            int kb = half * 8 + 2 * j + (j >> 2) * 8;
            bpk[j] = (unsigned int)f2bf(wrow[kk + kb]) | ((unsigned int)f2bf(wrow[kk + kb + 1]) << 16);
        }
        v16bf a0 = __builtin_bit_cast(v16bf, apk0);
        v16bf a1 = __builtin_bit_cast(v16bf, apk1);
        v16bf bf = __builtin_bit_cast(v16bf, bpk);
        acc0 = __builtin_amdgcn_wmma_f32_16x16x32_bf16(false, a0, false, bf, (short)0, acc0, false, false);
        acc1 = __builtin_amdgcn_wmma_f32_16x16x32_bf16(false, a1, false, bf, (short)0, acc1, false, false);
    }
    float bg = bgen[vcol];
    #pragma unroll
    for (int r = 0; r < 8; ++r) {
        int brw = half * 8 + r;
        out[(size_t)brw * VV + vcol] = acc0[r] + bg;
        out[(size_t)(brw + 16) * VV + vcol] = acc1[r] + bg;
    }
}

// ---------------- in-place log_softmax over each vocab row ----------------
__global__ __launch_bounds__(256) void lsm_kernel(float* __restrict__ base)
{
    int b = blockIdx.x, tid = threadIdx.x;          // 32 blocks x 256
    float* row = base + (size_t)b * VV;
    __shared__ float red[256];
    float m = -3.4e38f;
    for (int i = tid; i < VV; i += 256) m = fmaxf(m, row[i]);
    red[tid] = m; __syncthreads();
    for (int o = 128; o > 0; o >>= 1) { if (tid < o) red[tid] = fmaxf(red[tid], red[tid + o]); __syncthreads(); }
    float M = red[0]; __syncthreads();
    float s = 0.f;
    for (int i = tid; i < VV; i += 256) s += expf(row[i] - M);
    red[tid] = s; __syncthreads();
    for (int o = 128; o > 0; o >>= 1) { if (tid < o) red[tid] += red[tid + o]; __syncthreads(); }
    float lg = M + logf(red[0]);
    for (int i = tid; i < VV; i += 256) row[i] -= lg;
}

extern "C" void kernel_launch(void* const* d_in, const int* in_sizes, int n_in,
                              void* d_out, int out_size, void* d_ws, size_t ws_size,
                              hipStream_t stream) {
    (void)in_sizes; (void)n_in; (void)out_size;
    const int*   tgt   = (const int*)  d_in[0];   // (T,B,1)
    const float* mb    = (const float*)d_in[1];   // (S,B,H)
    const int*   lens  = (const int*)  d_in[2];   // (B,)
    const float* enc_h = (const float*)d_in[3];   // (2,B,H)
    const float* enc_c = (const float*)d_in[4];
    const float* emb   = (const float*)d_in[5];   // (V,E)
    const float* Wx0   = (const float*)d_in[6];
    const float* Wh0   = (const float*)d_in[7];
    const float* b0    = (const float*)d_in[8];
    const float* Wx1   = (const float*)d_in[9];
    const float* Wh1   = (const float*)d_in[10];
    const float* b1    = (const float*)d_in[11];
    const float* Wa    = (const float*)d_in[12];
    const float* Wc    = (const float*)d_in[13];
    const float* Wgen  = (const float*)d_in[14];
    const float* bgen  = (const float*)d_in[15];

    // workspace layout (float offsets)
    float* ws   = (float*)d_ws;
    float* h0   = ws;                    // B*H   } contiguous [h0|h1] for one D2D copy
    float* h1   = ws + 16384;            // B*H   }
    float* c0   = ws + 32768;            // B*H   } contiguous [c0|c1]
    float* c1   = ws + 49152;            // B*H   }
    float* feed = ws + 65536;            // B*H
    float* ctx  = ws + 81920;            // B*H
    float* q    = ws + 98304;            // B*H
    float* xc   = ws + 114688;           // B*(E+H) = 32768
    float* g0   = ws + 147456;           // B*4H
    float* g1   = ws + 212992;           // B*4H
    unsigned short* decbf  = (unsigned short*)(ws + 278528);   // B*H bf16
    unsigned short* wgenbf = (unsigned short*)(ws + 286720);   // V*H bf16 = 32.77MB

    const size_t need_big = (size_t)(286720 + (VV / 2) * HH) * sizeof(float);
    const bool big = ws_size >= need_big;

    hipMemcpyAsync(h0, enc_h, 2 * BB * HH * sizeof(float), hipMemcpyDeviceToDevice, stream);
    hipMemcpyAsync(c0, enc_c, 2 * BB * HH * sizeof(float), hipMemcpyDeviceToDevice, stream);
    hipMemsetAsync(feed, 0, BB * HH * sizeof(float), stream);
    if (big) cvt_wgen_kernel<<<VV * HH / 1024, 256, 0, stream>>>(Wgen, wgenbf);

    float* scores = (float*)d_out;                       // (NT,B,V)
    float* attns  = scores + (size_t)NT * BB * VV;       // (NT,B,S)

    for (int t = 0; t < NT; ++t) {
        const int* tgt_t = tgt + t * BB;
        xcat_kernel<<<128, 256, 0, stream>>>(tgt_t, emb, feed, xc);
        gates_wmma_kernel<1024, 512><<<NG / 64, 128, 0, stream>>>(xc, Wx0, h0, Wh0, b0, g0);
        cell0_kernel<<<64, 256, 0, stream>>>(g0, h0, c0);
        gates_wmma_kernel<512, 512><<<NG / 64, 128, 0, stream>>>(h0, Wx1, h1, Wh1, b1, g1);
        cell1q_kernel<<<32, 512, 0, stream>>>(g1, h1, c1, Wa, q);
        attn_kernel<<<32, 256, 0, stream>>>(q, mb, lens, attns + (size_t)t * BB * SS, ctx);
        dec_kernel<<<64, 256, 0, stream>>>(ctx, h1, Wc, feed, decbf);
        if (big)
            wgen_wmma_fast<<<VV / 64, 128, 0, stream>>>(decbf, wgenbf, bgen, scores + (size_t)t * BB * VV);
        else
            wgen_wmma_fly<<<VV / 64, 128, 0, stream>>>(decbf, Wgen, bgen, scores + (size_t)t * BB * VV);
        lsm_kernel<<<32, 256, 0, stream>>>(scores + (size_t)t * BB * VV);
    }
}
